// HandWritingSynthRNN_73340861547063
// MI455X (gfx1250) — matrix operand order repaired and checked by reading, not compiled, not run on hardware
//
#include <hip/hip_runtime.h>
#include <hip/hip_bf16.h>

// ---------------------------------------------------------------------------
// CDNA5 (gfx1250) WMMA types
// ---------------------------------------------------------------------------
typedef __attribute__((ext_vector_type(16))) _Float16 v16h;
typedef __attribute__((ext_vector_type(8)))  float    v8f;

#define T_STEPS 1000
#define BATCH   32
#define U_LEN   59
#define NCHAR   57
#define HID     400
#define GMIX    20
#define KATT    10

// Gate GEMM tiling: N = 4*HID = 1600 -> 100 N-tiles of 16.
#define NNT     100
// K layouts (all padded so recurrent h occupies k in [0,416) ):
//   layer0: [h(400) pad16 | x(3) win(57) pad4]      -> 480 = 15 k-tiles
//   layer1/2: [h(400) pad16 | hs_prev(400) x(3) win(57) pad20] -> 896 = 28 k-tiles
#define NKT0    15
#define NKT12   28
#define REC_PAD 416
// Final head GEMM: K = 3*HID = 1200 -> pad 1216 = 38 k-tiles, N = 121 -> pad 128 = 8 n-tiles.
#define NKTL    38
#define NNTL    8

#define SCAN_THREADS 800   // 25 waves; wave w owns h-units [16w,16w+16)

// d_out layout (flat, reference return order)
#define OFF_E    0
#define OFF_PI   32000
#define OFF_MU   672000
#define OFF_SIG  1952000
#define OFF_RHO  3232000
#define OFF_WIN  3872000
#define OFF_KAP  3873824

__device__ __forceinline__ float sigm(float x) { return 1.0f / (1.0f + __expf(-x)); }

__device__ __forceinline__ v8f wmma16x16x32(v16h a, v16h b, v8f c) {
  // D = A(16x32 f16) x B(32x16 f16) + C(16x16 f32)
  return __builtin_amdgcn_wmma_f32_16x16x32_f16(false, a, false, b, (short)0, c,
                                                false, false);
}

// A-fragment slot (16-bit A, 16x32 per ISA layout) for logical (batch row b, k).
// Lane = (b&15) + 16*((k%32 / 8) & 1);  element j packs (k&7) + 8*(k%32 / 16).
// Fragment storage: aF[(ktile*2 + b/16)*512 + lane*16 + j], v16h per lane.
__device__ __forceinline__ int afrag_slot(int b, int k) {
  int kt = k >> 5, kk = k & 31;
  int fl = (b & 15) + (((kk >> 3) & 1) << 4);
  int fj = (kk & 7) + ((kk >> 4) << 3);
  return (kt * 2 + (b >> 4)) * 512 + fl * 16 + fj;
}

// ---------------------------------------------------------------------------
// Weight swizzle: pack W (row-major [N][K]) as B-fragments (32x16 f16):
// lane l holds column n = nt*16 + (l&15), k = kt*32 + (l>=16 ? 8 : 0) + (j&7) + (j>=8 ? 16 : 0).
// k < recPad -> recurrent matrix (Krec real cols); k >= recPad -> input matrix.
// ---------------------------------------------------------------------------
__global__ void swizzle_weights(const float* __restrict__ Wrec, int Krec,
                                const float* __restrict__ Wih, int Kih,
                                int recPad, int Nreal, int numKt, int numNt,
                                _Float16* __restrict__ out) {
  int idx = blockIdx.x * blockDim.x + threadIdx.x;
  int total = numKt * numNt * 512;
  if (idx >= total) return;
  int frag = idx >> 9;
  int e    = idx & 511;
  int l = e >> 4, j = e & 15;
  int kt = frag / numNt, nt = frag % numNt;
  int n = nt * 16 + (l & 15);
  int k = kt * 32 + (j & 7) + ((j >> 3) << 4) + ((l >> 4) << 3);
  float v = 0.0f;
  if (n < Nreal) {
    if (k < recPad) {
      if (k < Krec) v = Wrec[n * Krec + k];
    } else {
      int ks = k - recPad;
      if (ks < Kih) v = Wih[n * Kih + ks];
    }
  }
  out[idx] = (_Float16)v;
}

// ---------------------------------------------------------------------------
// Layer 0: LSTM + attention window, persistent single-workgroup scan.
// ---------------------------------------------------------------------------
__global__ void __launch_bounds__(SCAN_THREADS)
scan_layer0(const float* __restrict__ inp, const float* __restrict__ c_seq,
            const _Float16* __restrict__ Bsw, const float* __restrict__ bias,
            const float* __restrict__ W_h2w, const float* __restrict__ b_h2w,
            float* __restrict__ hs0, float* __restrict__ wins,
            float* __restrict__ outWin, float* __restrict__ outKap) {
  __shared__ alignas(32) _Float16 aF[NKT0 * 1024];   // A fragments, 30.7 KB
  __shared__ float winS[BATCH * NCHAR];
  __shared__ float kapS[BATCH * KATT];
  __shared__ float wpS[BATCH * 30];
  __shared__ float phiS[BATCH * U_LEN];

  const int tid  = threadIdx.x;
  const int wave = tid >> 5;     // 0..24 == h-tile
  const int lane = tid & 31;

  for (int i = tid; i < NKT0 * 1024; i += SCAN_THREADS) aF[i] = (_Float16)0.0f;
  for (int i = tid; i < BATCH * NCHAR; i += SCAN_THREADS) winS[i] = 0.0f;
  for (int i = tid; i < BATCH * KATT; i += SCAN_THREADS) kapS[i] = 0.0f;

  const int u = wave * 16 + (lane & 15);
  const float bI = bias[u], bF = bias[HID + u], bG = bias[2 * HID + u],
              bO = bias[3 * HID + u];
  v8f cState[2];
  cState[0] = (v8f){};
  cState[1] = (v8f){};
  __syncthreads();

  for (int t = 0; t < T_STEPS; ++t) {
    // x_t into static fragment region (k = REC_PAD + 0..2)
    for (int idx = tid; idx < BATCH * 3; idx += SCAN_THREADS) {
      int b = idx / 3, kx = idx % 3;
      aF[afrag_slot(b, REC_PAD + kx)] = (_Float16)inp[(t * BATCH + b) * 3 + kx];
    }
    __syncthreads();

    // gates(32x1600) = [x|win|h] @ [W_ih0;W_hh0]^T   via WMMA.
    // Keep kt loop rolled: otherwise LICM hoists all loop-invariant weight
    // fragments out of the t-loop and spills them to scratch.
    v8f acc[4][2];
    {
      v8f z = {};
      for (int g = 0; g < 4; ++g) { acc[g][0] = z; acc[g][1] = z; }
    }
#pragma clang loop unroll(disable)
    for (int kt = 0; kt < NKT0; ++kt) {
      v16h a0 = *(const v16h*)(&aF[(kt * 2 + 0) * 512 + lane * 16]);
      v16h a1 = *(const v16h*)(&aF[(kt * 2 + 1) * 512 + lane * 16]);
#pragma unroll
      for (int g = 0; g < 4; ++g) {
        int nt = g * 25 + wave;
        v16h bf = *(const v16h*)(&Bsw[(size_t)(kt * NNT + nt) * 512 + lane * 16]);
        acc[g][0] = wmma16x16x32(a0, bf, acc[g][0]);
        acc[g][1] = wmma16x16x32(a1, bf, acc[g][1]);
      }
    }

    // LSTM cell entirely in accumulator registers; write h to LDS frags + global.
#pragma unroll
    for (int mt = 0; mt < 2; ++mt) {
#pragma unroll
      for (int r = 0; r < 8; ++r) {
        int b = mt * 16 + r + ((lane >> 4) << 3);
        float iv = acc[0][mt][r] + bI;
        float fv = acc[1][mt][r] + bF;
        float gv = acc[2][mt][r] + bG;
        float ov = acc[3][mt][r] + bO;
        float cn = sigm(fv) * cState[mt][r] + sigm(iv) * tanhf(gv);
        float hv = sigm(ov) * tanhf(cn);
        cState[mt][r] = cn;
        hs0[(size_t)(t * BATCH + b) * HID + u] = hv;
        aF[afrag_slot(b, u)] = (_Float16)hv;
      }
    }
    __syncthreads();

    // wp = exp(h @ W_h2w^T + b_h2w)   (32x30, VALU; h read back from frags)
    for (int idx = tid; idx < BATCH * 30; idx += SCAN_THREADS) {
      int b = idx / 30, jj = idx % 30;
      float s = b_h2w[jj];
      const float* wr = &W_h2w[jj * HID];
#pragma clang loop unroll_count(4)
      for (int k = 0; k < HID; ++k)
        s += (float)aF[afrag_slot(b, k)] * wr[k];
      wpS[idx] = __expf(s);
    }
    __syncthreads();

    // kappa += dkappa
    for (int idx = tid; idx < BATCH * KATT; idx += SCAN_THREADS) {
      int b = idx / KATT, kk = idx % KATT;
      kapS[idx] += wpS[b * 30 + 20 + kk];
    }
    __syncthreads();

    // phi[b,u] = sum_k alpha * exp(-beta*(kappa-u')^2)
    for (int idx = tid; idx < BATCH * U_LEN; idx += SCAN_THREADS) {
      int b = idx / U_LEN, uu = idx % U_LEN;
      float up = (float)(uu + 1);
      float s = 0.0f;
#pragma unroll
      for (int kk = 0; kk < KATT; ++kk) {
        float d = kapS[b * KATT + kk] - up;
        s += wpS[b * 30 + kk] * __expf(-wpS[b * 30 + 10 + kk] * d * d);
      }
      phiS[idx] = s;
    }
    __syncthreads();

    // win[b,c] = sum_u phi[b,u]*c_seq[b,u,c]; feed frags for next step
    for (int idx = tid; idx < BATCH * NCHAR; idx += SCAN_THREADS) {
      int b = idx / NCHAR, cc = idx % NCHAR;
      float s = 0.0f;
#pragma clang loop unroll(disable)
      for (int uu = 0; uu < U_LEN; ++uu)
        s += phiS[b * U_LEN + uu] * c_seq[(b * U_LEN + uu) * NCHAR + cc];
      winS[idx] = s;
      wins[(size_t)t * BATCH * NCHAR + idx] = s;
      aF[afrag_slot(b, REC_PAD + 3 + cc)] = (_Float16)s;
    }
    __syncthreads();
  }

  for (int idx = tid; idx < BATCH * NCHAR; idx += SCAN_THREADS) outWin[idx] = winS[idx];
  for (int idx = tid; idx < BATCH * KATT; idx += SCAN_THREADS) outKap[idx] = kapS[idx];
}

// ---------------------------------------------------------------------------
// Layers 1/2: plain LSTM scan, input = [h_rec | hs_prev | x | win].
// ---------------------------------------------------------------------------
__global__ void __launch_bounds__(SCAN_THREADS)
scan_layer12(const _Float16* __restrict__ Bsw, const float* __restrict__ bias,
             const float* __restrict__ hprev, const float* __restrict__ inp,
             const float* __restrict__ wins, float* __restrict__ hsOut) {
  __shared__ alignas(32) _Float16 aF[NKT12 * 1024];   // 57.3 KB

  const int tid  = threadIdx.x;
  const int wave = tid >> 5;
  const int lane = tid & 31;

  for (int i = tid; i < NKT12 * 1024; i += SCAN_THREADS) aF[i] = (_Float16)0.0f;

  const int u = wave * 16 + (lane & 15);
  const float bI = bias[u], bF = bias[HID + u], bG = bias[2 * HID + u],
              bO = bias[3 * HID + u];
  v8f cState[2];
  cState[0] = (v8f){};
  cState[1] = (v8f){};
  __syncthreads();

  for (int t = 0; t < T_STEPS; ++t) {
    // Build static fragments (k-tiles 13..27) from hs_prev / x / win.
    for (int idx = tid; idx < 15 * 1024; idx += SCAN_THREADS) {
      int kt  = 13 + (idx >> 10);
      int rem = idx & 1023;
      int mt = rem >> 9, e = rem & 511;
      int l = e >> 4, j = e & 15;
      int row = mt * 16 + (l & 15);
      int k = kt * 32 + (j & 7) + ((j >> 3) << 4) + ((l >> 4) << 3);
      int ks = k - REC_PAD;
      float v = 0.0f;
      if (ks < HID)               v = hprev[(size_t)(t * BATCH + row) * HID + ks];
      else if (ks < HID + 3)      v = inp[(t * BATCH + row) * 3 + (ks - HID)];
      else if (ks < HID + 3 + NCHAR)
                                  v = wins[(size_t)(t * BATCH + row) * NCHAR + (ks - HID - 3)];
      aF[kt * 1024 + mt * 512 + e] = (_Float16)v;
    }
    __syncthreads();

    v8f acc[4][2];
    {
      v8f z = {};
      for (int g = 0; g < 4; ++g) { acc[g][0] = z; acc[g][1] = z; }
    }
#pragma clang loop unroll(disable)
    for (int kt = 0; kt < NKT12; ++kt) {
      v16h a0 = *(const v16h*)(&aF[(kt * 2 + 0) * 512 + lane * 16]);
      v16h a1 = *(const v16h*)(&aF[(kt * 2 + 1) * 512 + lane * 16]);
#pragma unroll
      for (int g = 0; g < 4; ++g) {
        int nt = g * 25 + wave;
        v16h bf = *(const v16h*)(&Bsw[(size_t)(kt * NNT + nt) * 512 + lane * 16]);
        acc[g][0] = wmma16x16x32(a0, bf, acc[g][0]);
        acc[g][1] = wmma16x16x32(a1, bf, acc[g][1]);
      }
    }

#pragma unroll
    for (int mt = 0; mt < 2; ++mt) {
#pragma unroll
      for (int r = 0; r < 8; ++r) {
        int b = mt * 16 + r + ((lane >> 4) << 3);
        float iv = acc[0][mt][r] + bI;
        float fv = acc[1][mt][r] + bF;
        float gv = acc[2][mt][r] + bG;
        float ov = acc[3][mt][r] + bO;
        float cn = sigm(fv) * cState[mt][r] + sigm(iv) * tanhf(gv);
        float hv = sigm(ov) * tanhf(cn);
        cState[mt][r] = cn;
        hsOut[(size_t)(t * BATCH + b) * HID + u] = hv;
        aF[afrag_slot(b, u)] = (_Float16)hv;
      }
    }
    __syncthreads();
  }
}

// ---------------------------------------------------------------------------
// MDN head: y = [hs0|hs1|hs2] @ W_last^T + b_last, then softmax/exp/tanh/sig.
// One block per timestep; 8 waves = 8 N-tiles; A gathered straight from global.
// ---------------------------------------------------------------------------
__global__ void __launch_bounds__(256)
final_mdn(const float* __restrict__ hs0, const float* __restrict__ hs1,
          const float* __restrict__ hs2, const _Float16* __restrict__ Bsw,
          const float* __restrict__ b_last, float* __restrict__ out) {
  __shared__ float yS[BATCH * 128];
  const int t    = blockIdx.x;
  const int tid  = threadIdx.x;
  const int wave = tid >> 5;   // n-tile 0..7
  const int lane = tid & 31;

  v8f acc0 = {}, acc1 = {};
#pragma clang loop unroll(disable)
  for (int kt = 0; kt < NKTL; ++kt) {
    v16h a[2];
#pragma unroll
    for (int mt = 0; mt < 2; ++mt) {
      int row = mt * 16 + (lane & 15);
      int kb  = kt * 32 + ((lane >> 4) << 3);
#pragma unroll
      for (int half = 0; half < 2; ++half) {
        int k0 = kb + half * 16;    // 8 contiguous k, never straddles a source
        const float* src = nullptr;
        if (k0 < HID)          src = &hs0[(size_t)(t * BATCH + row) * HID + k0];
        else if (k0 < 2 * HID) src = &hs1[(size_t)(t * BATCH + row) * HID + (k0 - HID)];
        else if (k0 < 3 * HID) src = &hs2[(size_t)(t * BATCH + row) * HID + (k0 - 2 * HID)];
#pragma unroll
        for (int j = 0; j < 8; ++j)
          a[mt][half * 8 + j] = (_Float16)(src ? src[j] : 0.0f);
      }
    }
    v16h bf = *(const v16h*)(&Bsw[(size_t)(kt * NNTL + wave) * 512 + lane * 16]);
    acc0 = wmma16x16x32(a[0], bf, acc0);
    acc1 = wmma16x16x32(a[1], bf, acc1);
  }

#pragma unroll
  for (int mt = 0; mt < 2; ++mt) {
    v8f acc = mt ? acc1 : acc0;
#pragma unroll
    for (int r = 0; r < 8; ++r) {
      int b   = mt * 16 + r + ((lane >> 4) << 3);
      int col = wave * 16 + (lane & 15);
      float bb = (col < 121) ? b_last[col] : 0.0f;
      yS[b * 128 + col] = acc[r] + bb;
    }
  }
  __syncthreads();

  if (tid < BATCH) {
    int b = tid;
    const float* y = &yS[b * 128];
    int rg = t * BATCH + b;
    // pi = softmax(y[0:20])
    float m = y[0];
    for (int j = 1; j < GMIX; ++j) m = fmaxf(m, y[j]);
    float ex[GMIX];
    float sum = 0.0f;
    for (int j = 0; j < GMIX; ++j) { ex[j] = __expf(y[j] - m); sum += ex[j]; }
    float inv = 1.0f / sum;
    for (int j = 0; j < GMIX; ++j)     out[OFF_PI  + (size_t)rg * GMIX + j]     = ex[j] * inv;
    for (int j = 0; j < 2 * GMIX; ++j) out[OFF_MU  + (size_t)rg * 2 * GMIX + j] = y[GMIX + j];
    for (int j = 0; j < 2 * GMIX; ++j) out[OFF_SIG + (size_t)rg * 2 * GMIX + j] = __expf(y[3 * GMIX + j]);
    for (int j = 0; j < GMIX; ++j)     out[OFF_RHO + (size_t)rg * GMIX + j]     = tanhf(y[5 * GMIX + j]);
    out[OFF_E + rg] = sigm(y[6 * GMIX]);
  }
}

// ---------------------------------------------------------------------------
extern "C" void kernel_launch(void* const* d_in, const int* in_sizes, int n_in,
                              void* d_out, int out_size, void* d_ws, size_t ws_size,
                              hipStream_t stream) {
  (void)in_sizes; (void)n_in; (void)out_size; (void)ws_size;
  const float* inp    = (const float*)d_in[0];
  const float* c_seq  = (const float*)d_in[1];
  const float* W_ih0  = (const float*)d_in[2];
  const float* W_hh0  = (const float*)d_in[3];
  const float* b0     = (const float*)d_in[4];
  const float* W_ih1  = (const float*)d_in[5];
  const float* W_hh1  = (const float*)d_in[6];
  const float* b1     = (const float*)d_in[7];
  const float* W_ih2  = (const float*)d_in[8];
  const float* W_hh2  = (const float*)d_in[9];
  const float* b2     = (const float*)d_in[10];
  const float* W_h2w  = (const float*)d_in[11];
  const float* b_h2w  = (const float*)d_in[12];
  const float* W_last = (const float*)d_in[13];
  const float* b_last = (const float*)d_in[14];
  float* out = (float*)d_out;

  char* ws = (char*)d_ws;
  size_t off = 0;
  auto take = [&](size_t bytes) -> char* {
    char* p = ws + off;
    off = (off + bytes + 255) & ~(size_t)255;
    return p;
  };
  float* hs0  = (float*)take(sizeof(float) * (size_t)T_STEPS * BATCH * HID);
  float* hs1  = (float*)take(sizeof(float) * (size_t)T_STEPS * BATCH * HID);
  float* hs2  = (float*)take(sizeof(float) * (size_t)T_STEPS * BATCH * HID);
  float* wins = (float*)take(sizeof(float) * (size_t)T_STEPS * BATCH * NCHAR);
  _Float16* Bsw0 = (_Float16*)take(sizeof(_Float16) * (size_t)NKT0  * NNT  * 512);
  _Float16* Bsw1 = (_Float16*)take(sizeof(_Float16) * (size_t)NKT12 * NNT  * 512);
  _Float16* Bsw2 = (_Float16*)take(sizeof(_Float16) * (size_t)NKT12 * NNT  * 512);
  _Float16* BswL = (_Float16*)take(sizeof(_Float16) * (size_t)NKTL  * NNTL * 512);

  {
    int total = NKT0 * NNT * 512;
    swizzle_weights<<<(total + 255) / 256, 256, 0, stream>>>(
        W_hh0, HID, W_ih0, 3 + NCHAR, REC_PAD, 4 * HID, NKT0, NNT, Bsw0);
  }
  {
    int total = NKT12 * NNT * 512;
    swizzle_weights<<<(total + 255) / 256, 256, 0, stream>>>(
        W_hh1, HID, W_ih1, HID + 3 + NCHAR, REC_PAD, 4 * HID, NKT12, NNT, Bsw1);
    swizzle_weights<<<(total + 255) / 256, 256, 0, stream>>>(
        W_hh2, HID, W_ih2, HID + 3 + NCHAR, REC_PAD, 4 * HID, NKT12, NNT, Bsw2);
  }
  {
    int total = NKTL * NNTL * 512;
    swizzle_weights<<<(total + 255) / 256, 256, 0, stream>>>(
        W_last, 3 * HID, nullptr, 0, NKTL * 32, 6 * GMIX + 1, NKTL, NNTL, BswL);
  }

  scan_layer0<<<1, SCAN_THREADS, 0, stream>>>(inp, c_seq, Bsw0, b0, W_h2w, b_h2w,
                                              hs0, wins, out + OFF_WIN, out + OFF_KAP);
  scan_layer12<<<1, SCAN_THREADS, 0, stream>>>(Bsw1, b1, hs0, inp, wins, hs1);
  scan_layer12<<<1, SCAN_THREADS, 0, stream>>>(Bsw2, b2, hs1, inp, wins, hs2);
  final_mdn<<<T_STEPS, 256, 0, stream>>>(hs0, hs1, hs2, BswL, b_last, out);
}
